// BinarizeConv2dSDP_48945447305760
// MI455X (gfx1250) — 8-bit matrix operand order repaired and checked by reading, compiled, not run on hardware
//
#include <hip/hip_runtime.h>


typedef int v4i __attribute__((ext_vector_type(4)));
typedef int v8i __attribute__((ext_vector_type(8)));

#define IN_CHN  128
#define OUT_CHN 128
#define KS      3
#define NB      32
#define H_      56
#define W_      56
#define HP      58
#define WP      58
#define NW      147456
#define NCHUNK  18
#define KZ      5
#define SCALE_  100.0f

__global__ __launch_bounds__(256) void synth_pack_weights(
    const float* __restrict__ M, const float* __restrict__ Z,
    const float* __restrict__ rv, signed char* __restrict__ Apk)
{
    int i = blockIdx.x * blockDim.x + threadIdx.x;
    if (i >= NW) return;
    int b     = i & 3;
    int v     = (i >> 2) & 7;
    int lane  = (i >> 5) & 31;
    int chunk = (i >> 10) % NCHUNK;
    int ot    = i / (1024 * NCHUNK);

    int half = chunk & 1, tap = chunk >> 1;
    int kh = tap / 3, kw = tap % 3;
    int row = lane & 15, hi = lane >> 4;
    int kk  = (v >> 2) * 32 + hi * 16 + (v & 3) * 4 + b;
    int o   = ot * 16 + row;
    int c   = half * 64 + kk;
    int widx = ((o * IN_CHN + c) * KS + kh) * KS + kw;

    float m = M[widx];
    float a = m * m;
    float dot = 0.0f;
#pragma unroll
    for (int k = 0; k < KZ; ++k) {
        float zk = Z[k * NW + widx];
        a   += zk * zk * (1.0f / SCALE_);
        dot += rv[k] * zk;
    }
    float w = (m + dot) * rsqrtf(a);
    Apk[i] = (w > 0.0f) ? (signed char)1 : ((w < 0.0f) ? (signed char)-1 : (signed char)0);
}

__global__ __launch_bounds__(256) void binarize_pad(
    const float* __restrict__ x, unsigned int* __restrict__ act)
{
    int j = blockIdx.x * blockDim.x + threadIdx.x;
    const int total = NB * HP * WP * (IN_CHN / 4);
    if (j >= total) return;
    int c4 = j & 31;
    int xp = (j >> 5) % WP;
    int yp = (j / (32 * WP)) % HP;
    int n  =  j / (32 * WP * HP);

    unsigned int pack = 0u;
    if (yp >= 1 && yp <= H_ && xp >= 1 && xp <= W_) {
        int y = yp - 1, x0 = xp - 1;
#pragma unroll
        for (int bb = 0; bb < 4; ++bb) {
            int c = c4 * 4 + bb;
            float v = x[(((size_t)n * IN_CHN + c) * H_ + y) * W_ + x0];
            unsigned int sb = (v > 0.0f) ? 0x01u : ((v < 0.0f) ? 0xFFu : 0x00u);
            pack |= sb << (8 * bb);
        }
    }
    act[j] = pack;
}

__global__ __launch_bounds__(256) void bconv_wmma(
    const signed char* __restrict__ Apk,
    const signed char* __restrict__ act,
    const float* __restrict__ Alpha,
    float* __restrict__ out)
{
    const int lane = threadIdx.x & 31;
    const int wave = threadIdx.x >> 5;
    const int tile = blockIdx.x * 8 + wave;
    const int ot = tile & 7;
    const int pt = (tile >> 3) % 98;
    const int n  = tile / (8 * 98);

    const int col = lane & 15;
    const int hi  = lane >> 4;
    const int p0 = pt * 32 + col;
    const int p1 = p0 + 16;
    const int y0 = p0 / W_, x0 = p0 % W_;
    const int y1 = p1 / W_, x1 = p1 % W_;

    v8i acc0 = {};
    v8i acc1 = {};

    const signed char* apBase = Apk + (size_t)ot * (NCHUNK * 1024) + lane * 32;
    const signed char* actN   = act + (size_t)n * (HP * WP * IN_CHN);

#pragma unroll
    for (int tap = 0; tap < 9; ++tap) {
        const int kh = tap / 3, kw = tap % 3;
        const signed char* b0base =
            actN + (((y0 + kh) * WP + (x0 + kw)) * IN_CHN) + hi * 16;
        const signed char* b1base =
            actN + (((y1 + kh) * WP + (x1 + kw)) * IN_CHN) + hi * 16;
#pragma unroll
        for (int half = 0; half < 2; ++half) {
            const int chunk = tap * 2 + half;
            v8i a = *(const v8i*)(apBase + (size_t)chunk * 1024);
            v4i bl0 = *(const v4i*)(b0base + half * 64);
            v4i bh0 = *(const v4i*)(b0base + half * 64 + 32);
            v4i bl1 = *(const v4i*)(b1base + half * 64);
            v4i bh1 = *(const v4i*)(b1base + half * 64 + 32);
            v8i b0 = __builtin_shufflevector(bl0, bh0, 0, 1, 2, 3, 4, 5, 6, 7);
            v8i b1 = __builtin_shufflevector(bl1, bh1, 0, 1, 2, 3, 4, 5, 6, 7);
            acc0 = __builtin_amdgcn_wmma_i32_16x16x64_iu8(true, a, true, b0,
                                                          acc0, false, false);
            acc1 = __builtin_amdgcn_wmma_i32_16x16x64_iu8(true, a, true, b1,
                                                          acc1, false, false);
        }
    }

    float* outN = out + (size_t)n * OUT_CHN * (H_ * W_);
#pragma unroll
    for (int r = 0; r < 8; ++r) {
        const int o = ot * 16 + hi * 8 + r;
        const float av = Alpha[o];
        outN[(size_t)o * (H_ * W_) + p0] = (float)acc0[r] * av;
        outN[(size_t)o * (H_ * W_) + p1] = (float)acc1[r] * av;
    }
}

extern "C" void kernel_launch(void* const* d_in, const int* in_sizes, int n_in,
                              void* d_out, int out_size, void* d_ws, size_t ws_size,
                              hipStream_t stream)
{
    const float* x     = (const float*)d_in[0];
    const float* Alpha = (const float*)d_in[1];
    const float* Mw    = (const float*)d_in[2];
    const float* Zw    = (const float*)d_in[3];
    const float* rv    = (const float*)d_in[4];
    float* out = (float*)d_out;

    signed char* Apk = (signed char*)d_ws;
    signed char* act = (signed char*)d_ws + (size_t)262144;

    synth_pack_weights<<<(NW + 255) / 256, 256, 0, stream>>>(Mw, Zw, rv, Apk);

    {
        const int total = NB * HP * WP * (IN_CHN / 4);
        binarize_pad<<<(total + 255) / 256, 256, 0, stream>>>(
            x, (unsigned int*)act);
    }

    bconv_wmma<<<3136, 256, 0, stream>>>(Apk, act, Alpha, out);
}
